// DotProductAttention_16724602651348
// MI455X (gfx1250) — matrix operand order repaired and checked by reading, not compiled, not run on hardware
//
#include <hip/hip_runtime.h>

typedef __attribute__((ext_vector_type(16))) _Float16 v16h;
typedef __attribute__((ext_vector_type(8)))  _Float16 v8h;
typedef __attribute__((ext_vector_type(4)))  _Float16 v4h;
typedef __attribute__((ext_vector_type(8)))  float    v8f;

#define BH     64
#define N_CTX  2048
#define D_HEAD 64

union U16 { v16h v; v8h h[2]; };
union PU  { v16h v; unsigned int d[8]; };
union HP  { unsigned int u; _Float16 h[2]; };
union F8  { float4 f[2]; float s[8]; };

__device__ __forceinline__ v8f wmma32(v16h a, v16h b, v8f c) {
  // D = A(16x32 f16) * B(32x16 f16) + C(16x16 f32)
  return __builtin_amdgcn_wmma_f32_16x16x32_f16(false, a, false, b, (short)0, c,
                                                false, false);
}

__device__ __forceinline__ float fast_tanh(float x) {
#if __has_builtin(__builtin_amdgcn_tanhf)
  return __builtin_amdgcn_tanhf(x);            // v_tanh_f32
#else
  float e = exp2f(x * 2.8853900817779268f);
  return 1.f - 2.f / (e + 1.f);
#endif
}

__device__ __forceinline__ float fast_exp2(float x) {
  // args are in [-28.86, 0] here: raw v_exp_f32 is exact, skip libm's
  // subnormal-range fixup (cmp/cndmask/rescale chain).
#if __has_builtin(__builtin_amdgcn_exp2f)
  return __builtin_amdgcn_exp2f(x);            // v_exp_f32
#else
  return exp2f(x);
#endif
}

// ---------------- pass 1a: K fp32 -> f16 (same layout) ----------------
__global__ void cvt_f16_kernel(const float* __restrict__ src,
                               _Float16* __restrict__ dst, int n4) {
  int i = blockIdx.x * blockDim.x + threadIdx.x;
  if (i < n4) {
    float4 f = ((const float4*)src)[i];
    v4h r = {(_Float16)f.x, (_Float16)f.y, (_Float16)f.z, (_Float16)f.w};
    ((v4h*)dst)[i] = r;
  }
}

// ---------------- pass 1b: V fp32 [bh][n][d] -> f16 V^T [bh][d][n] ----------------
__global__ void vt_kernel(const float* __restrict__ V, _Float16* __restrict__ Vt) {
  __shared__ float tile[16][17];
  const int bh = blockIdx.z;
  const int n0 = blockIdx.x * 16, d0 = blockIdx.y * 16;
  const int tx = threadIdx.x, ty = threadIdx.y;
  tile[ty][tx] = V[((size_t)bh * N_CTX + n0 + ty) * D_HEAD + d0 + tx];
  __syncthreads();
  Vt[((size_t)bh * D_HEAD + d0 + ty) * N_CTX + n0 + tx] = (_Float16)tile[tx][ty];
}

// ---------------- pass 1c: bool mask -> f32 multiplicand (True -> 0.0) ----------------
__global__ void mexp_kernel(const unsigned char* __restrict__ m,
                            float* __restrict__ mf, int n) {
  int i = blockIdx.x * blockDim.x + threadIdx.x;
  if (i < n) mf[i] = m[i] ? 0.f : 1.f;
}

// ---------------- pass 2: flash attention, one wave per 16-query tile ----------------
// Transposed formulation: S^T = K x Q^T (Q pre-scaled by 1/8), O^T = V^T x P^T.
// tanh clip bounds logits to [-10,10] => fixed softmax max of 10:
// p = exp2((10*tanh(s) - 10)*log2e) * maskmul, in [e^-20, 1]. No running max,
// no rescale; row-sum reduced across the lane pair once at the end.
__global__ __launch_bounds__(256, 1)
void fattn_kernel(const float* __restrict__ Q, const _Float16* __restrict__ Kh,
                  const _Float16* __restrict__ Vt,
                  const float* __restrict__ Mf,
                  float* __restrict__ out) {
  const int lane  = threadIdx.x & 31;
  const int wave  = threadIdx.x >> 5;
  const int gtile = blockIdx.x * 8 + wave;   // 0..8191
  const int bh    = gtile >> 7;
  const int qt    = gtile & 127;
  const int b     = bh >> 3;
  const int qbase = qt * 16;
  const int qn    = lane & 15;               // query / column index of this lane
  const int hi    = lane >> 4;               // lane half

  // ---- Q^T B-fragments (pre-scaled by 1/sqrt(D)=1/8): lane = column(q);
  // halves = K(d) in contiguous 16-blocks per lane-half: d = df*32 + hi*16 + h
  const float* qp = Q + ((size_t)bh * N_CTX + qbase + qn) * D_HEAD;
  v16h bq[2];
#pragma unroll
  for (int df = 0; df < 2; ++df) {
    const float* p = qp + df * 32 + hi * 16;
    U16 u;
#pragma unroll
    for (int i = 0; i < 16; ++i) u.v[i] = (_Float16)(p[i] * 0.125f);
    bq[df] = u.v;
  }

  v8f o[4] = {};                 // O^T accumulators: d = dt*16 + v + 8*hi, col q
  float lsum = 0.f;              // lane-local partial row sum

  // running pointers (immediate offsets cover df/mt/dt within an iteration)
  const _Float16* kA = Kh + (size_t)bh * N_CTX * D_HEAD + (size_t)qn * D_HEAD + hi * 16;
  const _Float16* vA = Vt + (size_t)bh * D_HEAD * N_CTX + (size_t)qn * N_CTX + hi * 16;
  const float*    mA = Mf + (size_t)b * N_CTX + hi * 8;

  for (int kk = 0; kk < N_CTX; kk += 32) {
    __builtin_prefetch(kA + 32 * D_HEAD, 0, 3);
    __builtin_prefetch(vA + 32, 0, 3);

    // ---- S^T = K_tile x Q^T : A layout halves0-15: d=hi*16+0..15
    v8f c[2] = {};
#pragma unroll
    for (int mt = 0; mt < 2; ++mt) {
#pragma unroll
      for (int df = 0; df < 2; ++df) {
        const _Float16* kb = kA + mt * 16 * D_HEAD + df * 32;
        U16 u;
        u.h[0] = *(const v8h*)(kb);
        u.h[1] = *(const v8h*)(kb + 8);
        c[mt] = wmma32(u.v, bq[df], c[mt]);
      }
    }

    // ---- mask multiplicands for this lane's 16 keys (aligned b128 pairs)
    F8 mw0, mw1;
    mw0.f[0] = *(const float4*)(mA);      mw0.f[1] = *(const float4*)(mA + 4);
    mw1.f[0] = *(const float4*)(mA + 16); mw1.f[1] = *(const float4*)(mA + 20);

    // ---- p = exp2((10*tanh(s) - 10)*log2e) * maskmul
    float p0[8], p1[8];
#pragma unroll
    for (int v = 0; v < 8; ++v) {
      float t0 = fast_tanh(c[0][v]);
      float t1 = fast_tanh(c[1][v]);
      float e0 = fast_exp2(fmaf(t0, 14.4269504088896341f, -14.4269504088896341f)) * mw0.s[v];
      float e1 = fast_exp2(fmaf(t1, 14.4269504088896341f, -14.4269504088896341f)) * mw1.s[v];
      p0[v] = e0; p1[v] = e1;
      lsum += e0 + e1;
    }

    // ---- pack to f16 dwords, then swap lane halves (4 dword shuffles)
    unsigned int pk0[4], pk1[4], sel[4];
#pragma unroll
    for (int j = 0; j < 4; ++j) {
      HP a, bb;
      a.h[0]  = (_Float16)p0[2 * j];  a.h[1]  = (_Float16)p0[2 * j + 1];
      bb.h[0] = (_Float16)p1[2 * j];  bb.h[1] = (_Float16)p1[2 * j + 1];
      pk0[j] = a.u; pk1[j] = bb.u;
      sel[j] = __shfl_xor(hi ? pk0[j] : pk1[j], 16, 32);
    }
    // P^T B-fragment: hi=0: [own pk0 | partner pk0] ; hi=1: [partner pk1 | own pk1]
    PU pu;
#pragma unroll
    for (int j = 0; j < 4; ++j) {
      pu.d[j]     = hi ? sel[j] : pk0[j];
      pu.d[4 + j] = hi ? pk1[j] : sel[j];
    }

    // ---- O^T += V^T x P^T
#pragma unroll
    for (int dt = 0; dt < 4; ++dt) {
      const _Float16* vb = vA + (size_t)dt * 16 * N_CTX;
      U16 u;
      u.h[0] = *(const v8h*)(vb);
      u.h[1] = *(const v8h*)(vb + 8);
      o[dt] = wmma32(u.v, pu.v, o[dt]);
    }

    kA += 32 * D_HEAD;
    vA += 32;
    mA += 32;
  }

  // ---- finalize row sum across the lane pair, then store O = O^T/rsum
  float rsum = lsum + __shfl_xor(lsum, 16, 32);
  float inv  = 1.f / fmaxf(rsum, 1e-30f);
  float* op = out + ((size_t)bh * N_CTX + qbase + qn) * D_HEAD;
#pragma unroll
  for (int dt = 0; dt < 4; ++dt) {
    float4 w0 = {o[dt][0] * inv, o[dt][1] * inv, o[dt][2] * inv, o[dt][3] * inv};
    float4 w1 = {o[dt][4] * inv, o[dt][5] * inv, o[dt][6] * inv, o[dt][7] * inv};
    float* dst = op + dt * 16 + hi * 8;
    *(float4*)dst       = w0;
    *(float4*)(dst + 4) = w1;
  }
}

extern "C" void kernel_launch(void* const* d_in, const int* in_sizes, int n_in,
                              void* d_out, int out_size, void* d_ws, size_t ws_size,
                              hipStream_t stream) {
  const float*         Q    = (const float*)d_in[0];
  const float*         K    = (const float*)d_in[1];
  const float*         V    = (const float*)d_in[2];
  const unsigned char* mask = (const unsigned char*)d_in[3]; // jnp.bool_ = 1 byte/elem
  float*               out  = (float*)d_out;

  _Float16* Kh = (_Float16*)d_ws;                   // 16.8 MB
  _Float16* Vt = Kh + (size_t)BH * N_CTX * D_HEAD;  // 16.8 MB
  float*    Mf = (float*)(Vt + (size_t)BH * N_CTX * D_HEAD); // 64 KB (ws >= ~34 MB)

  const int n  = BH * N_CTX * D_HEAD;               // 8,388,608
  const int n4 = n / 4;
  cvt_f16_kernel<<<(n4 + 255) / 256, 256, 0, stream>>>(K, Kh, n4);
  vt_kernel<<<dim3(N_CTX / 16, D_HEAD / 16, BH), dim3(16, 16), 0, stream>>>(V, Vt);
  mexp_kernel<<<(8 * N_CTX + 255) / 256, 256, 0, stream>>>(mask, Mf, 8 * N_CTX);

  const int tiles = BH * (N_CTX / 16);              // 8192 wave-tiles, 8 waves/block
  fattn_kernel<<<tiles / 8, 256, 0, stream>>>(Q, Kh, Vt, Mf, out);
}
